// DrugProteinAttention_58574763983244
// MI455X (gfx1250) — compile-verified, matrix operand order checked by reading, not run on hardware
//
#include <hip/hip_runtime.h>
#include <hip/hip_bf16.h>

// Problem constants (fixed by the reference).
#define N_NODES 20000
#define N_EDGES 320000
#define DIN     256
#define DMODEL  128
#define NHEAD   4
#define HEADD   32
#define NTILE   8                         // DMODEL / 16 output column tiles
#define ATTN_SCALE 0.17677669529663687f   // 1/sqrt(32)

typedef __attribute__((ext_vector_type(16))) __bf16 v16bf;
typedef __attribute__((ext_vector_type(8)))  __bf16 v8bf;
typedef __attribute__((ext_vector_type(8)))  float  v8f;

// -------- helpers --------
static __device__ __forceinline__ __bf16 f2bf(float f) {
  unsigned u = __float_as_uint(f);
  u += 0x7FFFu + ((u >> 16) & 1u);          // round-to-nearest-even
  unsigned short h = (unsigned short)(u >> 16);
  return __builtin_bit_cast(__bf16, h);
}

static __device__ __forceinline__ void atomicMaxFloat(float* addr, float val) {
  unsigned int* ua = (unsigned int*)addr;
  unsigned int old = *ua;
  while (__uint_as_float(old) < val) {
    unsigned int assumed = old;
    old = atomicCAS(ua, assumed, __float_as_uint(val));
    if (old == assumed) break;
  }
}

static __device__ __forceinline__ v16bf load_frag(const __bf16* base, int ka) {
  v8bf c0 = *(const v8bf*)(base + ka);
  v8bf c1 = *(const v8bf*)(base + ka + 16);
  v16bf f;
#pragma unroll
  for (int i = 0; i < 8; ++i) { f[i] = c0[i]; f[i + 8] = c1[i]; }
  return f;
}

// -------- elementwise kernels --------
__global__ void cvt_f32_to_bf16(const float* __restrict__ src,
                                __bf16* __restrict__ dst, int n) {
  int i = blockIdx.x * blockDim.x + threadIdx.x;
  if (i < n) dst[i] = f2bf(src[i]);
}

__global__ void fill_f32(float* __restrict__ p, float v, int n) {
  int i = blockIdx.x * blockDim.x + threadIdx.x;
  if (i < n) p[i] = v;
}

// -------- WMMA GEMM strip kernel --------
// C[M x 128] = alpha * (A[M x K](bf16) @ W[128 x K]^T(bf16) + bias)
// One wave per 16-row strip: A fragment loaded ONCE per K-step, reused across the
// 8 column tiles (8 independent WMMA accumulation chains -> good XDL pipelining,
// 8x less activation traffic than a tile-per-wave scheme).
// CDNA5 bf16 A/B fragment layout: row = lane&15; per-lane K chunks
// [klo, klo+7] and [klo+16, klo+23] with klo = (lane>>4)*8.
// C/D f32 layout: VGPR r, lane l -> (m = r + 8*(l>=16), n = l&15).
__global__ void __launch_bounds__(32)
gemm_bf16_wmma(const __bf16* __restrict__ A, const __bf16* __restrict__ W,
               const float* __restrict__ bias, float alpha,
               float* __restrict__ Cf, __bf16* __restrict__ Cbf,
               int M, int K) {
  const int lane  = threadIdx.x;
  const int mbase = blockIdx.x * 16;
  const int row   = lane & 15;          // A row / W row within a tile
  const int hi    = lane >> 4;          // 0 or 1
  const int klo0  = hi * 8;

  v8f acc[NTILE];
#pragma unroll
  for (int nt = 0; nt < NTILE; ++nt) {
    const float b = bias ? bias[nt * 16 + row] : 0.0f;
#pragma unroll
    for (int r = 0; r < 8; ++r) acc[nt][r] = b;
  }

  const __bf16* arow  = A + (size_t)(mbase + row) * K;
  const __bf16* wbase = W + (size_t)row * K;   // row within each 16-row W tile

  for (int k0 = 0; k0 < K; k0 += 32) {
    const int ka = k0 + klo0;
    const v16bf av = load_frag(arow, ka);
#pragma unroll
    for (int nt = 0; nt < NTILE; ++nt) {
      const v16bf bv = load_frag(wbase + (size_t)(nt * 16) * K, ka);
      acc[nt] = __builtin_amdgcn_wmma_f32_16x16x32_bf16(
          /*neg_a=*/false, av, /*neg_b=*/false, bv,
          /*c_mod=*/(short)0, acc[nt], /*reuse_a=*/false, /*reuse_b=*/false);
    }
  }

#pragma unroll
  for (int nt = 0; nt < NTILE; ++nt) {
#pragma unroll
    for (int r = 0; r < 8; ++r) {
      const float val = acc[nt][r] * alpha;
      const int m = mbase + r + hi * 8;
      const int n = nt * 16 + row;
      const size_t idx = (size_t)m * DMODEL + n;
      if (Cf)  Cf[idx]  = val;
      if (Cbf) Cbf[idx] = f2bf(val);
    }
  }
}

// -------- edge phase 1: logits + segment max (one wave per edge) --------
// HEADD == warpSize == 32: each of the 4 head-dots reduces across the full wave.
__global__ void __launch_bounds__(256)
edge_logits_max(const int* __restrict__ qidx, const int* __restrict__ kidx,
                const float* __restrict__ q, const float* __restrict__ k,
                float* __restrict__ logits, float* __restrict__ mmax, int nE) {
  const int e    = blockIdx.x * (blockDim.x >> 5) + (threadIdx.x >> 5);
  const int lane = threadIdx.x & 31;
  if (e >= nE) return;                       // whole wave exits together
  const int qi = qidx[e];
  const int ki = kidx[e];
  const float* qp = q + (size_t)qi * DMODEL;
  const float* kp = k + (size_t)ki * DMODEL;
  float s[NHEAD];
#pragma unroll
  for (int h = 0; h < NHEAD; ++h)
    s[h] = qp[h * HEADD + lane] * kp[h * HEADD + lane];
#pragma unroll
  for (int off = 16; off > 0; off >>= 1) {
#pragma unroll
    for (int h = 0; h < NHEAD; ++h) s[h] += __shfl_xor(s[h], off, 32);
  }
  if (lane < NHEAD) {                        // 4 lanes write 4 heads in parallel
    const float lg = s[lane];
    logits[(size_t)e * NHEAD + lane] = lg;
    atomicMaxFloat(&mmax[(size_t)qi * NHEAD + lane], lg);
  }
}

// -------- edge phase 2: exp + segment sums (128 threads per edge) --------
__global__ void __launch_bounds__(256)
edge_accum(const int* __restrict__ qidx, const int* __restrict__ kidx,
           const float* __restrict__ logits, const float* __restrict__ mmax,
           const float* __restrict__ v,
           float* __restrict__ num, float* __restrict__ den, int nE) {
  const int e = blockIdx.x * 2 + (threadIdx.x >> 7);
  const int t = threadIdx.x & 127;
  if (e >= nE) return;
  const int qi = qidx[e];
  const int ki = kidx[e];
  const int h  = t >> 5;
  const float ex = __expf(logits[(size_t)e * NHEAD + h] -
                          mmax[(size_t)qi * NHEAD + h]);
  atomicAdd(&num[(size_t)qi * DMODEL + t], ex * v[(size_t)ki * DMODEL + t]);
  if ((t & 31) == 0) atomicAdd(&den[(size_t)qi * NHEAD + h], ex);
}

// -------- attn = num/denom, emit bf16 for the output GEMM --------
__global__ void attn_normalize(const float* __restrict__ num,
                               const float* __restrict__ den,
                               __bf16* __restrict__ attn_bf, int total) {
  int i = blockIdx.x * blockDim.x + threadIdx.x;
  if (i >= total) return;
  const int node = i >> 7;
  const int h    = (i & 127) >> 5;
  const float d  = den[(size_t)node * NHEAD + h];
  attn_bf[i] = f2bf(num[i] / fmaxf(d, 1e-30f));
}

// -------- select(has_edge, out, h_q) + ReLU --------
__global__ void epilogue_relu(const float* __restrict__ outf,
                              const float* __restrict__ den,
                              const float* __restrict__ hq,
                              float* __restrict__ dst, int total) {
  int i = blockIdx.x * blockDim.x + threadIdx.x;
  if (i >= total) return;
  const int node = i >> 7;
  const float val = (den[(size_t)node * NHEAD] > 0.0f) ? outf[i] : hq[i];
  dst[i] = fmaxf(val, 0.0f);
}

extern "C" void kernel_launch(void* const* d_in, const int* in_sizes, int n_in,
                              void* d_out, int out_size, void* d_ws, size_t ws_size,
                              hipStream_t stream) {
  const float* x_src    = (const float*)d_in[0];
  const float* x_dst    = (const float*)d_in[1];
  const int*   edge     = (const int*)d_in[2];
  const float* w_src    = (const float*)d_in[3];
  const float* w_dst    = (const float*)d_in[4];
  const float* inw_s2d  = (const float*)d_in[5];
  const float* inb_s2d  = (const float*)d_in[6];
  const float* outw_s2d = (const float*)d_in[7];
  const float* outb_s2d = (const float*)d_in[8];
  const float* inw_d2s  = (const float*)d_in[9];
  const float* inb_d2s  = (const float*)d_in[10];
  const float* outw_d2s = (const float*)d_in[11];
  const float* outb_d2s = (const float*)d_in[12];

  const int* src_ids = edge;             // edge_index[0]
  const int* dst_ids = edge + N_EDGES;   // edge_index[1]

  // ---- workspace carve-out (256B aligned) ----
  char* ws = (char*)d_ws;
  size_t off = 0;
  auto alloc = [&](size_t bytes) -> char* {
    char* p = ws + off;
    off += (bytes + 255) & ~(size_t)255;
    return p;
  };
  const size_t ND  = (size_t)N_NODES * DMODEL;
  const size_t NDI = (size_t)N_NODES * DIN;

  __bf16* xs_bf     = (__bf16*)alloc(NDI * 2);
  __bf16* xd_bf     = (__bf16*)alloc(NDI * 2);
  __bf16* wsrc_bf   = (__bf16*)alloc((size_t)DMODEL * DIN * 2);
  __bf16* wdst_bf   = (__bf16*)alloc((size_t)DMODEL * DIN * 2);
  __bf16* inws_bf   = (__bf16*)alloc((size_t)3 * DMODEL * DMODEL * 2);
  __bf16* inwd_bf   = (__bf16*)alloc((size_t)3 * DMODEL * DMODEL * 2);
  __bf16* outws_bf  = (__bf16*)alloc((size_t)DMODEL * DMODEL * 2);
  __bf16* outwd_bf  = (__bf16*)alloc((size_t)DMODEL * DMODEL * 2);
  float*  h_src_f   = (float*)alloc(ND * 4);
  __bf16* h_src_bf  = (__bf16*)alloc(ND * 2);
  float*  h_dst_f   = (float*)alloc(ND * 4);
  __bf16* h_dst_bf  = (__bf16*)alloc(ND * 2);
  float*  q_s2d = (float*)alloc(ND * 4);
  float*  k_s2d = (float*)alloc(ND * 4);
  float*  v_s2d = (float*)alloc(ND * 4);
  float*  q_d2s = (float*)alloc(ND * 4);
  float*  k_d2s = (float*)alloc(ND * 4);
  float*  v_d2s = (float*)alloc(ND * 4);
  float*  logits_s2d = (float*)alloc((size_t)N_EDGES * NHEAD * 4);
  float*  logits_d2s = (float*)alloc((size_t)N_EDGES * NHEAD * 4);
  float*  m_s2d   = (float*)alloc((size_t)N_NODES * NHEAD * 4);
  float*  m_d2s   = (float*)alloc((size_t)N_NODES * NHEAD * 4);
  float*  den_s2d = (float*)alloc((size_t)N_NODES * NHEAD * 4);
  float*  den_d2s = (float*)alloc((size_t)N_NODES * NHEAD * 4);
  float*  num_s2d = (float*)alloc(ND * 4);
  float*  num_d2s = (float*)alloc(ND * 4);
  __bf16* attn_s2d_bf = (__bf16*)alloc(ND * 2);
  __bf16* attn_d2s_bf = (__bf16*)alloc(ND * 2);
  float*  out_s2d_f = (float*)alloc(ND * 4);
  float*  out_d2s_f = (float*)alloc(ND * 4);

  auto cvt = [&](const float* s, __bf16* d, int n) {
    cvt_f32_to_bf16<<<(n + 255) / 256, 256, 0, stream>>>(s, d, n);
  };
  auto fill = [&](float* p, float v, int n) {
    fill_f32<<<(n + 255) / 256, 256, 0, stream>>>(p, v, n);
  };
  auto gemm = [&](const __bf16* A, const __bf16* W, const float* bias, float alpha,
                  float* Cf, __bf16* Cbf, int M, int K) {
    gemm_bf16_wmma<<<M / 16, 32, 0, stream>>>(A, W, bias, alpha, Cf, Cbf, M, K);
  };

  // ---- stage 0: bf16 conversion of activations & weights ----
  cvt(x_src, xs_bf, (int)NDI);
  cvt(x_dst, xd_bf, (int)NDI);
  cvt(w_src, wsrc_bf, DMODEL * DIN);
  cvt(w_dst, wdst_bf, DMODEL * DIN);
  cvt(inw_s2d, inws_bf, 3 * DMODEL * DMODEL);
  cvt(inw_d2s, inwd_bf, 3 * DMODEL * DMODEL);
  cvt(outw_s2d, outws_bf, DMODEL * DMODEL);
  cvt(outw_d2s, outwd_bf, DMODEL * DMODEL);

  // ---- stage 1: input projections (WMMA) ----
  gemm(xs_bf, wsrc_bf, nullptr, 1.0f, h_src_f, h_src_bf, N_NODES, DIN);
  gemm(xd_bf, wdst_bf, nullptr, 1.0f, h_dst_f, h_dst_bf, N_NODES, DIN);

  // ---- stage 2: QKV projections (WMMA). s2d: q<-h_dst, k/v<-h_src; d2s swapped ----
  const int DD = DMODEL * DMODEL;
  gemm(h_dst_bf, inws_bf,          inb_s2d,              ATTN_SCALE, q_s2d, nullptr, N_NODES, DMODEL);
  gemm(h_src_bf, inws_bf + DD,     inb_s2d + DMODEL,     1.0f,       k_s2d, nullptr, N_NODES, DMODEL);
  gemm(h_src_bf, inws_bf + 2 * DD, inb_s2d + 2 * DMODEL, 1.0f,       v_s2d, nullptr, N_NODES, DMODEL);
  gemm(h_src_bf, inwd_bf,          inb_d2s,              ATTN_SCALE, q_d2s, nullptr, N_NODES, DMODEL);
  gemm(h_dst_bf, inwd_bf + DD,     inb_d2s + DMODEL,     1.0f,       k_d2s, nullptr, N_NODES, DMODEL);
  gemm(h_dst_bf, inwd_bf + 2 * DD, inb_d2s + 2 * DMODEL, 1.0f,       v_d2s, nullptr, N_NODES, DMODEL);

  // ---- stage 3: segment softmax state init ----
  fill(m_s2d, -3.402823466e38f, N_NODES * NHEAD);
  fill(m_d2s, -3.402823466e38f, N_NODES * NHEAD);
  fill(den_s2d, 0.0f, N_NODES * NHEAD);
  fill(den_d2s, 0.0f, N_NODES * NHEAD);
  fill(num_s2d, 0.0f, (int)ND);
  fill(num_d2s, 0.0f, (int)ND);

  // ---- stage 4: edge phase ----
  edge_logits_max<<<(N_EDGES + 7) / 8, 256, 0, stream>>>(
      dst_ids, src_ids, q_s2d, k_s2d, logits_s2d, m_s2d, N_EDGES);
  edge_logits_max<<<(N_EDGES + 7) / 8, 256, 0, stream>>>(
      src_ids, dst_ids, q_d2s, k_d2s, logits_d2s, m_d2s, N_EDGES);
  edge_accum<<<(N_EDGES + 1) / 2, 256, 0, stream>>>(
      dst_ids, src_ids, logits_s2d, m_s2d, v_s2d, num_s2d, den_s2d, N_EDGES);
  edge_accum<<<(N_EDGES + 1) / 2, 256, 0, stream>>>(
      src_ids, dst_ids, logits_d2s, m_d2s, v_d2s, num_d2s, den_d2s, N_EDGES);

  // ---- stage 5: normalize + output projection (WMMA) + epilogue ----
  attn_normalize<<<((int)ND + 255) / 256, 256, 0, stream>>>(num_s2d, den_s2d, attn_s2d_bf, (int)ND);
  attn_normalize<<<((int)ND + 255) / 256, 256, 0, stream>>>(num_d2s, den_d2s, attn_d2s_bf, (int)ND);

  gemm(attn_s2d_bf, outws_bf, outb_s2d, 1.0f, out_s2d_f, nullptr, N_NODES, DMODEL);
  gemm(attn_d2s_bf, outwd_bf, outb_d2s, 1.0f, out_d2s_f, nullptr, N_NODES, DMODEL);

  float* out_upd_src = (float*)d_out;           // d2s pass (queries = h_src)
  float* out_upd_dst = (float*)d_out + ND;      // s2d pass (queries = h_dst)
  epilogue_relu<<<((int)ND + 255) / 256, 256, 0, stream>>>(out_d2s_f, den_d2s, h_src_f, out_upd_src, (int)ND);
  epilogue_relu<<<((int)ND + 255) / 256, 256, 0, stream>>>(out_s2d_f, den_s2d, h_dst_f, out_upd_dst, (int)ND);
}
